// LlamaRALAAttention_35175782154296
// MI455X (gfx1250) — matrix operand order repaired and checked by reading, not compiled, not run on hardware
//
#include <hip/hip_runtime.h>

typedef long long i64;

// ---------------------------------------------------------------------------
// Problem constants (match reference)
// ---------------------------------------------------------------------------
#define B_    2
#define S_    4096
#define HID_  2048
#define NH_   32
#define NKV_  8
#define HD_   64
#define GRP_  (NH_ / NKV_)

typedef __attribute__((ext_vector_type(16))) __bf16 v16bf;
typedef __attribute__((ext_vector_type(8)))  float  v8f;

union BFrag {
  v16bf v;
  uint4 q[2];
  unsigned short u[16];
};

__device__ __forceinline__ unsigned short f2bf(float f) {
  unsigned int u = __builtin_bit_cast(unsigned int, f);
  unsigned int r = u + 0x7FFFu + ((u >> 16) & 1u);   // round-to-nearest-even
  return (unsigned short)(r >> 16);
}
__device__ __forceinline__ float bf2f(unsigned short h) {
  unsigned int u = ((unsigned int)h) << 16;
  return __builtin_bit_cast(float, u);
}
// packed bf16x2 in a u32 -> two floats
__device__ __forceinline__ float blo(unsigned int u) { return __builtin_bit_cast(float, u << 16); }
__device__ __forceinline__ float bhi(unsigned int u) { return __builtin_bit_cast(float, u & 0xFFFF0000u); }

// ---------------------------------------------------------------------------
// f32 -> bf16 convert (grid-stride)
// ---------------------------------------------------------------------------
__global__ __launch_bounds__(256)
void cvt_bf16(const float* __restrict__ in, unsigned short* __restrict__ out, i64 n) {
  i64 i = (i64)blockIdx.x * blockDim.x + threadIdx.x;
  const i64 stride = (i64)gridDim.x * blockDim.x;
  for (; i < n; i += stride) out[i] = f2bf(in[i]);
}

// ---------------------------------------------------------------------------
// f32 [K,N] -> bf16 transposed [N,K]  (tiled transpose through LDS)
// ---------------------------------------------------------------------------
__global__ __launch_bounds__(256)
void cvt_bf16_tr(const float* __restrict__ in, unsigned short* __restrict__ out,
                 int K, int N) {
  __shared__ float t[32][33];
  const int n0 = blockIdx.x * 32, k0 = blockIdx.y * 32;
  const int tx = threadIdx.x, ty = threadIdx.y;
#pragma unroll
  for (int j = 0; j < 32; j += 8)
    t[ty + j][tx] = in[(i64)(k0 + ty + j) * N + (n0 + tx)];
  __syncthreads();
#pragma unroll
  for (int j = 0; j < 32; j += 8)
    out[(i64)(n0 + ty + j) * K + (k0 + tx)] = f2bf(t[tx][ty + j]);
}

// ---------------------------------------------------------------------------
// Tiled bf16 WMMA GEMM:  C[M,N] = A[M,K] * Bt[N,K]^T   (A row-major, B given
// pre-transposed).  Block tile 128x256, K-step 32, double-buffered LDS.
// 256 threads = 8 waves (2 in M x 4 in N); wave tile 64x64 = 16 wmma tiles.
// All fragment reads are contiguous ds_load_b128 pairs.
// ---------------------------------------------------------------------------
#define LDT 40   // padded LDS row stride (elements) for 32-K tiles

template <bool OUT_F32>
__global__ __launch_bounds__(256)
void gemm_bf16_128x256(const unsigned short* __restrict__ A,
                       const unsigned short* __restrict__ Bt,
                       void* __restrict__ C, int M, int N, int K) {
  __shared__ __align__(16) unsigned short sA[2][128 * LDT];
  __shared__ __align__(16) unsigned short sB[2][256 * LDT];

  const int tid  = threadIdx.x;
  const int lane = tid & 31;
  const int wave = tid >> 5;
  const int wm   = (wave & 1) * 64;   // wave M offset within 128
  const int wn   = (wave >> 1) * 64;  // wave N offset within 256
  const i64 bm   = (i64)blockIdx.x * 128;
  const i64 bn   = (i64)blockIdx.y * 256;
  const int lm   = lane & 15;
  const int lh   = lane >> 4;

  const v8f vz = {0.f, 0.f, 0.f, 0.f, 0.f, 0.f, 0.f, 0.f};
  v8f acc[4][4];
#pragma unroll
  for (int i = 0; i < 4; ++i)
#pragma unroll
    for (int j = 0; j < 4; ++j) acc[i][j] = vz;

  // loader mapping: A tile 128x32 -> 16 elems/thread; B tile 256x32 -> 32/thread
  const int arow = tid >> 1, acol = (tid & 1) * 16;
  const int brow = tid >> 1, bk   = (tid & 1) * 16;

  const unsigned short* gA  = A  + (bm + arow) * (i64)K + acol;
  const unsigned short* gB0 = Bt + (bn + brow) * (i64)K + bk;
  const unsigned short* gB1 = Bt + (bn + brow + 128) * (i64)K + bk;

  // prologue: fill buffer 0
  {
    const uint4* pa = reinterpret_cast<const uint4*>(gA);
    const uint4* p0 = reinterpret_cast<const uint4*>(gB0);
    const uint4* p1 = reinterpret_cast<const uint4*>(gB1);
    uint4 a0 = pa[0], a1 = pa[1];
    uint4 b0 = p0[0], b1 = p0[1], b2 = p1[0], b3 = p1[1];
    *reinterpret_cast<uint4*>(&sA[0][arow * LDT + acol])     = a0;
    *reinterpret_cast<uint4*>(&sA[0][arow * LDT + acol + 8]) = a1;
    *reinterpret_cast<uint4*>(&sB[0][brow * LDT + bk])       = b0;
    *reinterpret_cast<uint4*>(&sB[0][brow * LDT + bk + 8])   = b1;
    *reinterpret_cast<uint4*>(&sB[0][(brow + 128) * LDT + bk])     = b2;
    *reinterpret_cast<uint4*>(&sB[0][(brow + 128) * LDT + bk + 8]) = b3;
  }
  __syncthreads();

  int buf = 0;
  for (int k0 = 0; k0 < K; k0 += 32) {
    const bool pref = (k0 + 32) < K;
    uint4 a0, a1, b0, b1, b2, b3;
    if (pref) {  // issue next-tile global loads early (overlap with WMMA)
      const uint4* pa = reinterpret_cast<const uint4*>(gA + k0 + 32);
      const uint4* p0 = reinterpret_cast<const uint4*>(gB0 + k0 + 32);
      const uint4* p1 = reinterpret_cast<const uint4*>(gB1 + k0 + 32);
      a0 = pa[0]; a1 = pa[1];
      b0 = p0[0]; b1 = p0[1]; b2 = p1[0]; b3 = p1[1];
    }

    // ---- compute on current buffer ----
    BFrag bfr[4];
#pragma unroll
    for (int nt = 0; nt < 4; ++nt) {
      const int nc = wn + nt * 16 + lm;
      bfr[nt].q[0] = *reinterpret_cast<const uint4*>(&sB[buf][nc * LDT + lh * 16]);
      bfr[nt].q[1] = *reinterpret_cast<const uint4*>(&sB[buf][nc * LDT + lh * 16 + 8]);
    }
#pragma unroll
    for (int mt = 0; mt < 4; ++mt) {
      BFrag a;
      const int mr = wm + mt * 16 + lm;
      a.q[0] = *reinterpret_cast<const uint4*>(&sA[buf][mr * LDT + lh * 16]);
      a.q[1] = *reinterpret_cast<const uint4*>(&sA[buf][mr * LDT + lh * 16 + 8]);
#pragma unroll
      for (int nt = 0; nt < 4; ++nt)
        acc[mt][nt] = __builtin_amdgcn_wmma_f32_16x16x32_bf16(
            false, a.v, false, bfr[nt].v, (short)0, acc[mt][nt], false, false);
    }

    if (pref) {
      const int nb = buf ^ 1;
      *reinterpret_cast<uint4*>(&sA[nb][arow * LDT + acol])     = a0;
      *reinterpret_cast<uint4*>(&sA[nb][arow * LDT + acol + 8]) = a1;
      *reinterpret_cast<uint4*>(&sB[nb][brow * LDT + bk])       = b0;
      *reinterpret_cast<uint4*>(&sB[nb][brow * LDT + bk + 8])   = b1;
      *reinterpret_cast<uint4*>(&sB[nb][(brow + 128) * LDT + bk])     = b2;
      *reinterpret_cast<uint4*>(&sB[nb][(brow + 128) * LDT + bk + 8]) = b3;
    }
    __syncthreads();
    buf ^= 1;
  }

#pragma unroll
  for (int mt = 0; mt < 4; ++mt)
#pragma unroll
    for (int nt = 0; nt < 4; ++nt)
#pragma unroll
      for (int r = 0; r < 8; ++r) {
        const i64 m = bm + wm + mt * 16 + r + lh * 8;
        const i64 n = bn + wn + nt * 16 + lm;
        if (OUT_F32)
          reinterpret_cast<float*>(C)[m * N + n] = acc[mt][nt][r];
        else
          reinterpret_cast<unsigned short*>(C)[m * N + n] = f2bf(acc[mt][nt][r]);
      }
}

// ---------------------------------------------------------------------------
// RoPE cos/sin table:  [S, 32]  (identical for every head/batch)
// ---------------------------------------------------------------------------
__global__ __launch_bounds__(256)
void rope_table(float* __restrict__ cosT, float* __restrict__ sinT) {
  const int gid = blockIdx.x * 256 + threadIdx.x;   // S_*32 threads
  const int d = gid & 31, s = gid >> 5;
  const float invf = powf(10000.0f, -(float)d * (1.0f / 32.0f));
  float sv, cv;
  sincosf((float)s * invf, &sv, &cv);
  cosT[s * 32 + d] = cv;
  sinT[s * 32 + d] = sv;
}

// ---------------------------------------------------------------------------
// RoPE + kappa for Q, [B,S,NH*HD] -> [B,NH,S,HD].  One thread per (d, d+32)
// pair (cos/sin identical for the two halves).
// ---------------------------------------------------------------------------
__global__ __launch_bounds__(256)
void rope_kappa_q(const unsigned short* __restrict__ qproj,
                  const float* __restrict__ cosT, const float* __restrict__ sinT,
                  unsigned short* __restrict__ Qk) {
  const i64 gid = (i64)blockIdx.x * blockDim.x + threadIdx.x;  // B*NH*S*32
  const int d = (int)(gid & 31);
  i64 t = gid >> 5;
  const int s = (int)(t % S_); t /= S_;
  const int n = (int)(t % NH_);
  const int b = (int)(t / NH_);

  const i64 row = ((i64)b * S_ + s) * (i64)HID_ + (i64)n * HD_;
  const float x1 = bf2f(qproj[row + d]);
  const float x2 = bf2f(qproj[row + d + 32]);
  const float c  = cosT[s * 32 + d];
  const float sv = sinT[s * 32 + d];
  const float y1 = x1 * c - x2 * sv;
  const float y2 = x2 * c + x1 * sv;
  const float k1 = (y1 > 0.0f) ? (y1 + 1.0f) : expf(y1);
  const float k2 = (y2 > 0.0f) ? (y2 + 1.0f) : expf(y2);
  const i64 o = (((i64)b * NH_ + n) * S_ + s) * HD_ + d;
  Qk[o]      = f2bf(k1);
  Qk[o + 32] = f2bf(k2);
}

// ---------------------------------------------------------------------------
// RoPE + kappa for K and transpose of V:  -> [B,NKV,S,HD]
// ---------------------------------------------------------------------------
__global__ __launch_bounds__(256)
void rope_kappa_k_tr_v(const unsigned short* __restrict__ kproj,
                       const unsigned short* __restrict__ vproj,
                       const float* __restrict__ cosT, const float* __restrict__ sinT,
                       unsigned short* __restrict__ Kk,
                       unsigned short* __restrict__ V) {
  const i64 gid = (i64)blockIdx.x * blockDim.x + threadIdx.x;  // B*NKV*S*32
  const int d = (int)(gid & 31);
  i64 t = gid >> 5;
  const int s = (int)(t % S_); t /= S_;
  const int n = (int)(t % NKV_);
  const int b = (int)(t / NKV_);

  const i64 row = ((i64)b * S_ + s) * (i64)(NKV_ * HD_) + (i64)n * HD_;
  const float x1 = bf2f(kproj[row + d]);
  const float x2 = bf2f(kproj[row + d + 32]);
  const float c  = cosT[s * 32 + d];
  const float sv = sinT[s * 32 + d];
  const float y1 = x1 * c - x2 * sv;
  const float y2 = x2 * c + x1 * sv;
  const float k1 = (y1 > 0.0f) ? (y1 + 1.0f) : expf(y1);
  const float k2 = (y2 > 0.0f) ? (y2 + 1.0f) : expf(y2);
  const i64 o = (((i64)b * NKV_ + n) * S_ + s) * HD_ + d;
  Kk[o]      = f2bf(k1);
  Kk[o + 32] = f2bf(k2);
  V[o]       = vproj[row + d];
  V[o + 32]  = vproj[row + d + 32];
}

// ---------------------------------------------------------------------------
// Qg[b,n,d] = mean_s Qk[b,n,s,d]     (one block per (b,n))
// ---------------------------------------------------------------------------
__global__ __launch_bounds__(256)
void qg_reduce(const unsigned short* __restrict__ Qk, float* __restrict__ Qg) {
  const int bh  = blockIdx.x;
  const int tid = threadIdx.x;
  const int d   = tid & 63;
  const int q   = tid >> 6;
  const unsigned short* base = Qk + ((i64)bh * S_) * HD_;
  float acc = 0.f;
  for (int s = q * 1024; s < (q + 1) * 1024; ++s)
    acc += bf2f(base[(i64)s * HD_ + d]);
  __shared__ float red[256];
  red[tid] = acc;
  __syncthreads();
  if (q == 0)
    Qg[(i64)bh * HD_ + d] =
        (red[d] + red[64 + d] + red[128 + d] + red[192 + d]) * (1.0f / (float)S_);
}

// ---------------------------------------------------------------------------
// logits + softmax:  alpha[b,n,s] = softmax_s(Qg . Kk) * S   (block per (b,n))
// ---------------------------------------------------------------------------
__global__ __launch_bounds__(256)
void alpha_softmax(const unsigned short* __restrict__ Kk,
                   const float* __restrict__ Qg, float* __restrict__ alpha) {
  const int bh = blockIdx.x;
  const int b  = bh / NH_;
  const int n  = bh % NH_;
  const int kv = n / GRP_;
  const int tid = threadIdx.x;

  __shared__ float qg[HD_];
  __shared__ float red[256];
  if (tid < HD_) qg[tid] = Qg[(i64)bh * HD_ + tid];
  __syncthreads();

  const unsigned short* kbase = Kk + (((i64)b * NKV_ + kv) * S_) * HD_;
  float lg[16];
  float mx = -3.0e38f;
#pragma unroll
  for (int j = 0; j < 16; ++j) {
    const int s = j * 256 + tid;
    const uint4* kq = reinterpret_cast<const uint4*>(kbase + (i64)s * HD_);
    float acc = 0.f;
#pragma unroll
    for (int c = 0; c < 8; ++c) {
      const uint4 w = kq[c];
      acc += qg[c * 8 + 0] * blo(w.x) + qg[c * 8 + 1] * bhi(w.x);
      acc += qg[c * 8 + 2] * blo(w.y) + qg[c * 8 + 3] * bhi(w.y);
      acc += qg[c * 8 + 4] * blo(w.z) + qg[c * 8 + 5] * bhi(w.z);
      acc += qg[c * 8 + 6] * blo(w.w) + qg[c * 8 + 7] * bhi(w.w);
    }
    lg[j] = acc;
    mx = fmaxf(mx, acc);
  }
  red[tid] = mx;
  __syncthreads();
  for (int off = 128; off > 0; off >>= 1) {
    if (tid < off) red[tid] = fmaxf(red[tid], red[tid + off]);
    __syncthreads();
  }
  const float gmax = red[0];
  __syncthreads();

  float sum = 0.f;
#pragma unroll
  for (int j = 0; j < 16; ++j) {
    lg[j] = expf(lg[j] - gmax);
    sum += lg[j];
  }
  red[tid] = sum;
  __syncthreads();
  for (int off = 128; off > 0; off >>= 1) {
    if (tid < off) red[tid] += red[tid + off];
    __syncthreads();
  }
  const float scale = (float)S_ / red[0];
#pragma unroll
  for (int j = 0; j < 16; ++j)
    alpha[(i64)bh * S_ + j * 256 + tid] = lg[j] * scale;
}

// ---------------------------------------------------------------------------
// outerT[b,n,f,d] = sum_s alpha[s] * Kk[s,d] * V[s,f]   (TRANSPOSED store)
// = (alpha*Kk)^T @ V per head, 64x64x4096 WMMA GEMM; alpha folded into A-load.
// Block per (b,n): 128 threads = 4 waves.
// ---------------------------------------------------------------------------
__global__ __launch_bounds__(128)
void outer_accum(const unsigned short* __restrict__ Kk,
                 const unsigned short* __restrict__ V,
                 const float* __restrict__ alpha,
                 unsigned short* __restrict__ outerT) {
  const int bh = blockIdx.x;
  const int b  = bh / NH_;
  const int n  = bh % NH_;
  const int kv = n / GRP_;
  const int tid  = threadIdx.x;
  const int lane = tid & 31;
  const int wave = tid >> 5;
  const int lm = lane & 15;
  const int lh = lane >> 4;

  __shared__ __align__(16) unsigned short sK[32 * 64];
  __shared__ __align__(16) unsigned short sV[32 * 64];
  __shared__ float sAl[32];

  const unsigned short* kb = Kk + (((i64)b * NKV_ + kv) * S_) * HD_;
  const unsigned short* vb = V  + (((i64)b * NKV_ + kv) * S_) * HD_;
  const float* ab = alpha + (i64)bh * S_;

  const int row = tid >> 2, col = (tid & 3) * 16;
  const v8f vz = {0.f, 0.f, 0.f, 0.f, 0.f, 0.f, 0.f, 0.f};
  v8f acc[4] = {vz, vz, vz, vz};

  for (int s0 = 0; s0 < S_; s0 += 32) {
    const uint4* gk = reinterpret_cast<const uint4*>(kb + (i64)(s0 + row) * HD_ + col);
    const uint4* gv = reinterpret_cast<const uint4*>(vb + (i64)(s0 + row) * HD_ + col);
    uint4 k0v = gk[0], k1v = gk[1], v0v = gv[0], v1v = gv[1];
    *reinterpret_cast<uint4*>(&sK[row * 64 + col])     = k0v;
    *reinterpret_cast<uint4*>(&sK[row * 64 + col + 8]) = k1v;
    *reinterpret_cast<uint4*>(&sV[row * 64 + col])     = v0v;
    *reinterpret_cast<uint4*>(&sV[row * 64 + col + 8]) = v1v;
    if (tid < 32) sAl[tid] = ab[s0 + tid];
    __syncthreads();

    BFrag af;   // A[m=d][k=s] = alpha[s] * Kk[s,d]
    const int md = wave * 16 + lm;
#pragma unroll
    for (int i = 0; i < 16; ++i) {
      const int kk = (lh << 4) + i;
      af.u[i] = f2bf(bf2f(sK[kk * 64 + md]) * sAl[kk]);
    }
#pragma unroll
    for (int nt = 0; nt < 4; ++nt) {
      BFrag bfr;  // B[k=s][n=f] = V[s,f]
      const int nc = nt * 16 + lm;
      const int kbse = lh * 16;
#pragma unroll
      for (int i = 0; i < 16; ++i) bfr.u[i] = sV[(kbse + i) * 64 + nc];
      acc[nt] = __builtin_amdgcn_wmma_f32_16x16x32_bf16(
          false, af.v, false, bfr.v, (short)0, acc[nt], false, false);
    }
    __syncthreads();
  }

#pragma unroll
  for (int nt = 0; nt < 4; ++nt)
#pragma unroll
    for (int r = 0; r < 8; ++r) {
      const int m = wave * 16 + r + lh * 8;   // d
      const int f = nt * 16 + lm;             // f
      outerT[(i64)bh * 4096 + f * 64 + m] = f2bf(acc[nt][r]);  // [f][d]
    }
}

// ---------------------------------------------------------------------------
// result_attn = Qk @ outer  (per head, 4096x64x64), fused with X_phi Hadamard
// product -> ctx bf16 [B,S,HID].  outerT is [f][d] so B frags are contiguous.
// Grid: (B*NH, S/128); 256 threads = 8 waves, each wave a 16x64 strip.
// ---------------------------------------------------------------------------
#define LDO 72   // padded LDS stride for 64-wide outer tile

__global__ __launch_bounds__(256)
void attn_ctx(const unsigned short* __restrict__ Qk,
              const unsigned short* __restrict__ outerT,
              const unsigned short* __restrict__ xphi,
              unsigned short* __restrict__ ctx) {
  const int bh = blockIdx.x;
  const int b  = bh / NH_;
  const int n  = bh % NH_;
  const int tid  = threadIdx.x;
  const int lane = tid & 31;
  const int wave = tid >> 5;
  const int lm = lane & 15;
  const int lh = lane >> 4;

  __shared__ __align__(16) unsigned short sO[64 * LDO];  // [f][d], padded
  {
    const int row = tid >> 2, col = (tid & 3) * 16;
    const uint4* go = reinterpret_cast<const uint4*>(outerT + (i64)bh * 4096 + row * 64 + col);
    uint4 o0 = go[0], o1 = go[1];
    *reinterpret_cast<uint4*>(&sO[row * LDO + col])     = o0;
    *reinterpret_cast<uint4*>(&sO[row * LDO + col + 8]) = o1;
  }
  __syncthreads();

  const int s0 = blockIdx.y * 128 + wave * 16;
  const unsigned short* qb = Qk + ((i64)bh * S_) * HD_ + (i64)(s0 + lm) * HD_;

  const v8f vz = {0.f, 0.f, 0.f, 0.f, 0.f, 0.f, 0.f, 0.f};
  v8f acc[4] = {vz, vz, vz, vz};

#pragma unroll
  for (int kt = 0; kt < 2; ++kt) {
    BFrag af;   // A[m=s][k=d] = Qk[s0+m, d] : contiguous global b128 pair
    af.q[0] = *reinterpret_cast<const uint4*>(qb + kt * 32 + lh * 16);
    af.q[1] = *reinterpret_cast<const uint4*>(qb + kt * 32 + lh * 16 + 8);
#pragma unroll
    for (int nt = 0; nt < 4; ++nt) {
      BFrag bfr;  // B[k=d][n=f] = outerT[f][d] : contiguous LDS b128 pair
      const int nc = nt * 16 + lm;
      const int kbse = kt * 32 + lh * 16;
      bfr.q[0] = *reinterpret_cast<const uint4*>(&sO[nc * LDO + kbse]);
      bfr.q[1] = *reinterpret_cast<const uint4*>(&sO[nc * LDO + kbse + 8]);
      acc[nt] = __builtin_amdgcn_wmma_f32_16x16x32_bf16(
          false, af.v, false, bfr.v, (short)0, acc[nt], false, false);
    }
  }

#pragma unroll
  for (int nt = 0; nt < 4; ++nt)
#pragma unroll
    for (int r = 0; r < 8; ++r) {
      const int s = s0 + r + lh * 8;
      const int f = nt * 16 + lm;
      const i64 idx = ((i64)b * S_ + s) * (i64)HID_ + (i64)n * HD_ + f;
      ctx[idx] = f2bf(acc[nt][r] * bf2f(xphi[idx]));
    }
}

// ---------------------------------------------------------------------------
// Host-side launcher
// ---------------------------------------------------------------------------
extern "C" void kernel_launch(void* const* d_in, const int* in_sizes, int n_in,
                              void* d_out, int out_size, void* d_ws, size_t ws_size,
                              hipStream_t stream) {
  (void)in_sizes; (void)n_in; (void)out_size; (void)ws_size;
  const float* hs   = (const float*)d_in[0];
  const float* Wq   = (const float*)d_in[1];
  const float* Wk   = (const float*)d_in[2];
  const float* Wv   = (const float*)d_in[3];
  const float* Wphi = (const float*)d_in[4];
  const float* Wo   = (const float*)d_in[5];
  float* out = (float*)d_out;

  char* ws = (char*)d_ws;
  size_t off = 0;
  auto a16 = [&](size_t elems) -> unsigned short* {
    unsigned short* p = (unsigned short*)(ws + off);
    off += ((elems * 2 + 255) & ~(size_t)255);
    return p;
  };
  auto a32 = [&](size_t elems) -> float* {
    float* p = (float*)(ws + off);
    off += ((elems * 4 + 255) & ~(size_t)255);
    return p;
  };

  const size_t E_HS = (size_t)B_ * S_ * HID_;
  const size_t E_KV = (size_t)B_ * NKV_ * S_ * HD_;

  unsigned short* hsb    = a16(E_HS);
  unsigned short* wqb    = a16((size_t)HID_ * HID_);         // transposed [N,K]
  unsigned short* wkb    = a16((size_t)HID_ * NKV_ * HD_);
  unsigned short* wvb    = a16((size_t)HID_ * NKV_ * HD_);
  unsigned short* wphib  = a16((size_t)HID_ * HID_);
  unsigned short* wob    = a16((size_t)HID_ * HID_);
  unsigned short* qproj  = a16(E_HS);
  unsigned short* kproj  = a16(E_KV);
  unsigned short* vproj  = a16(E_KV);
  unsigned short* xphi   = a16(E_HS);
  unsigned short* Qk     = a16(E_HS);
  unsigned short* Kkb    = a16(E_KV);
  unsigned short* Vb     = a16(E_KV);
  unsigned short* outerT = a16((size_t)B_ * NH_ * HD_ * HD_);
  float* cosT  = a32((size_t)S_ * 32);
  float* sinT  = a32((size_t)S_ * 32);
  float* Qg    = a32((size_t)B_ * NH_ * HD_);
  float* alpha = a32((size_t)B_ * NH_ * S_);
  unsigned short* ctx = hsb;   // alias: hs_bf16 dead after projection GEMMs

  const dim3 blk(256);
  const int BSr = B_ * S_;     // 8192 rows

  // 1) converts (weights transposed to [N,K])
  cvt_bf16<<<4096, blk, 0, stream>>>(hs, hsb, (i64)E_HS);
  cvt_bf16_tr<<<dim3(HID_ / 32, HID_ / 32), dim3(32, 8), 0, stream>>>(Wq,   wqb,   HID_, HID_);
  cvt_bf16_tr<<<dim3(NKV_ * HD_ / 32, HID_ / 32), dim3(32, 8), 0, stream>>>(Wk, wkb, HID_, NKV_ * HD_);
  cvt_bf16_tr<<<dim3(NKV_ * HD_ / 32, HID_ / 32), dim3(32, 8), 0, stream>>>(Wv, wvb, HID_, NKV_ * HD_);
  cvt_bf16_tr<<<dim3(HID_ / 32, HID_ / 32), dim3(32, 8), 0, stream>>>(Wphi, wphib, HID_, HID_);
  cvt_bf16_tr<<<dim3(HID_ / 32, HID_ / 32), dim3(32, 8), 0, stream>>>(Wo,   wob,   HID_, HID_);
  rope_table<<<(S_ * 32) / 256, blk, 0, stream>>>(cosT, sinT);

  // 2) projection GEMMs (bf16 out)
  gemm_bf16_128x256<false><<<dim3(64, 8), blk, 0, stream>>>(hsb, wqb,   qproj, BSr, HID_,       HID_);
  gemm_bf16_128x256<false><<<dim3(64, 2), blk, 0, stream>>>(hsb, wkb,   kproj, BSr, NKV_ * HD_, HID_);
  gemm_bf16_128x256<false><<<dim3(64, 2), blk, 0, stream>>>(hsb, wvb,   vproj, BSr, NKV_ * HD_, HID_);
  gemm_bf16_128x256<false><<<dim3(64, 8), blk, 0, stream>>>(hsb, wphib, xphi,  BSr, HID_,       HID_);

  // 3) RoPE + kappa (+ transposes)
  rope_kappa_q<<<(unsigned)((E_HS / 2) / 256), blk, 0, stream>>>(qproj, cosT, sinT, Qk);
  rope_kappa_k_tr_v<<<(unsigned)((E_KV / 2) / 256), blk, 0, stream>>>(kproj, vproj, cosT, sinT, Kkb, Vb);

  // 4) global query + softmax
  qg_reduce<<<B_ * NH_, blk, 0, stream>>>(Qk, Qg);
  alpha_softmax<<<B_ * NH_, blk, 0, stream>>>(Kkb, Qg, alpha);

  // 5) KV state (per-head 64x64x4096 WMMA GEMM), transposed store
  outer_accum<<<B_ * NH_, dim3(128), 0, stream>>>(Kkb, Vb, alpha, outerT);

  // 6) result_attn fused with X_phi -> ctx
  attn_ctx<<<dim3(B_ * NH_, S_ / 128), blk, 0, stream>>>(Qk, outerT, xphi, ctx);

  // 7) final projection (f32 out)
  gemm_bf16_128x256<true><<<dim3(64, 8), blk, 0, stream>>>(ctx, wob, out, BSr, HID_, HID_);
}
